// GraphSAGE_44409961841194
// MI455X (gfx1250) — matrix operand order repaired and checked by reading, not compiled, not run on hardware
//
#include <hip/hip_runtime.h>
#include <stdint.h>

// ---------------------------------------------------------------------------
// CDNA5 / gfx1250 GraphSAGE forward.  wave32; WMMA f16->f32 for all GEMMs.
// Weights + activations pre-packed into native WMMA fragment layouts; packed
// activation buffers are padded to a multiple of MT row tiles so the GEMM
// inner loop is branch-free (pure b128 loads + v_wmma, fixed accumulators).
// ---------------------------------------------------------------------------

typedef _Float16 v16h __attribute__((ext_vector_type(16)));
typedef float    v8f  __attribute__((ext_vector_type(8)));

#define BN_EPS 1e-5f
#define MT 4  // row tiles per wave in the GEMM

// ----------------------------- utility fills -------------------------------
__global__ void fill_f32(float* __restrict__ p, float v, long long n) {
  long long i = (long long)blockIdx.x * blockDim.x + threadIdx.x;
  long long stride = (long long)gridDim.x * blockDim.x;
  for (; i < n; i += stride) p[i] = v;
}

// ------------------------------ degree -------------------------------------
__global__ void deg_kernel(const long long* __restrict__ dst,
                           float* __restrict__ deg, int E) {
  int i = blockIdx.x * blockDim.x + threadIdx.x;
  if (i < E) atomicAdd(&deg[(int)dst[i]], 1.0f);
}

__global__ void rdeg_kernel(float* __restrict__ deg, int N) {
  int i = blockIdx.x * blockDim.x + threadIdx.x;
  if (i < N) deg[i] = 1.0f / fmaxf(deg[i], 1.0f);
}

// --------------------------- fragment packing ------------------------------
// A-fragment layout (16x32 f16 tile, ISA 7.12.2): for element (m,k):
//   rt=m>>4, mr=m&15, kc=k>>5, kk=k&31, t=kk>>1, p=kk&1,
//   v=((t>>3)<<2)|(t&3), half=(t>>2)&1, lane=half*16+mr, j=2v+p
//   idx = ((rt*KC+kc)*32 + lane)*16 + j
__device__ __forceinline__ long long a_frag_idx(int m, int k, int KC) {
  int rt = m >> 4, mr = m & 15;
  int kc = k >> 5, kk = k & 31;
  int t = kk >> 1, p = kk & 1;
  int v = ((t >> 3) << 2) | (t & 3);
  int half = (t >> 2) & 1;
  int lane = half * 16 + mr;
  int j = v * 2 + p;
  return ((long long)(rt * KC + kc) * 32 + lane) * 16 + j;
}

// pack X[N,K] f32 (optionally row-scaled) -> A-fragment f16
__global__ void pack_a(const float* __restrict__ X,
                       const float* __restrict__ rowscale,
                       _Float16* __restrict__ out, int N, int K) {
  long long total = (long long)N * K;
  long long i = (long long)blockIdx.x * blockDim.x + threadIdx.x;
  long long stride = (long long)gridDim.x * blockDim.x;
  int KC = K >> 5;
  for (; i < total; i += stride) {
    int m = (int)(i / K), k = (int)(i % K);
    float s = rowscale ? rowscale[m] : 1.0f;
    out[a_frag_idx(m, k, KC)] = (_Float16)(X[i] * s);
  }
}

// pack W[Co,K] f32 -> B-fragment f16 for computing X @ W^T.
// B layout: same as A-fragment with n as the row: a_frag_idx(n, k, KC)
__global__ void pack_b(const float* __restrict__ W, _Float16* __restrict__ out,
                       int Co, int K) {
  long long total = (long long)Co * K;
  long long i = (long long)blockIdx.x * blockDim.x + threadIdx.x;
  long long stride = (long long)gridDim.x * blockDim.x;
  int KC = K >> 5;
  for (; i < total; i += stride) {
    int n = (int)(i / K), k = (int)(i % K);
    long long idx = a_frag_idx(n, k, KC);
    out[idx] = (_Float16)W[i];
  }
}

// ------------------------------ batchnorm ----------------------------------
__global__ void bn_stats(const float* __restrict__ x, float* __restrict__ stats,
                         int N, int F) {
  int f = threadIdx.x;  // blockDim.x == F
  float s = 0.0f, ss = 0.0f;
  for (int r = blockIdx.x; r < N; r += gridDim.x) {
    float v = x[(long long)r * F + f];
    s += v;
    ss += v * v;
  }
  atomicAdd(&stats[f], s);
  atomicAdd(&stats[F + f], ss);
}

// normalize; write f32 copy (for scatter) AND packed f16 A-fragment copy
__global__ void bn_apply_pack(const float* __restrict__ x,
                              const float* __restrict__ stats,
                              const float* __restrict__ g,
                              const float* __restrict__ b,
                              float* __restrict__ y,
                              _Float16* __restrict__ yh, int N, int F) {
  long long total = (long long)N * F;
  long long i = (long long)blockIdx.x * blockDim.x + threadIdx.x;
  long long stride = (long long)gridDim.x * blockDim.x;
  float rn = 1.0f / (float)N;
  int KC = F >> 5;
  for (; i < total; i += stride) {
    int m = (int)(i / F), f = (int)(i % F);
    float mean = stats[f] * rn;
    float var = stats[F + f] * rn - mean * mean;
    float v = (x[i] - mean) * rsqrtf(var + BN_EPS) * g[f] + b[f];
    y[i] = v;
    yh[a_frag_idx(m, f, KC)] = (_Float16)v;
  }
}

// --------------------------- neighbor scatter ------------------------------
__global__ void scatter_accum(const float* __restrict__ x,
                              const long long* __restrict__ src,
                              const long long* __restrict__ dst,
                              float* __restrict__ aggr, int E, int F) {
  int chunks = F >> 2;
  long long total = (long long)E * chunks;
  long long t = (long long)blockIdx.x * blockDim.x + threadIdx.x;
  long long stride = (long long)gridDim.x * blockDim.x;
  for (; t < total; t += stride) {
    int e = (int)(t / chunks);
    int c = (int)(t % chunks) << 2;
    int s = (int)src[e];
    int d = (int)dst[e];
    const float4 v = *(const float4*)(x + (long long)s * F + c);
    float* o = aggr + (long long)d * F + c;
    atomicAdd(o + 0, v.x);
    atomicAdd(o + 1, v.y);
    atomicAdd(o + 2, v.z);
    atomicAdd(o + 3, v.w);
  }
}

// ------------------------------ WMMA GEMM ----------------------------------
// Y[N,Co] = act( A1 @ W1^T (+ A2 @ W2^T if DUAL) + bias ).
// Packed-fragment inputs; A buffers are PADDED to a multiple of MT row tiles,
// so the k-loop has no guards and accumulators never move.  One wave per
// 16-col tile x MT row tiles.  4 (8) independent wmma chains per k-chunk.
template <bool DUAL>
__global__ __launch_bounds__(32) void wmma_gemm(
    const _Float16* __restrict__ A1, const _Float16* __restrict__ B1,
    const _Float16* __restrict__ A2, const _Float16* __restrict__ B2,
    const float* __restrict__ bias, float* __restrict__ Y,
    int N, int K, int Co, int relu) {
  int lane = threadIdx.x;
  int ct = blockIdx.x;
  int rt0 = blockIdx.y * MT;
  int KC = K >> 5;

  v8f c[MT] = {};
  for (int kc = 0; kc < KC; ++kc) {
    const v16h b1 =
        *(const v16h*)(B1 + ((long long)(ct * KC + kc) * 32 + lane) * 16);
    v16h a1[MT];
#pragma unroll
    for (int t = 0; t < MT; ++t)
      a1[t] = *(const v16h*)(A1 +
                             ((long long)((rt0 + t) * KC + kc) * 32 + lane) *
                                 16);
#pragma unroll
    for (int t = 0; t < MT; ++t)
      c[t] = __builtin_amdgcn_wmma_f32_16x16x32_f16(false, a1[t], false, b1,
                                                    (short)0, c[t], false,
                                                    false);
    if (DUAL) {
      const v16h b2 =
          *(const v16h*)(B2 + ((long long)(ct * KC + kc) * 32 + lane) * 16);
      v16h a2[MT];
#pragma unroll
      for (int t = 0; t < MT; ++t)
        a2[t] = *(const v16h*)(A2 +
                               ((long long)((rt0 + t) * KC + kc) * 32 + lane) *
                                   16);
#pragma unroll
      for (int t = 0; t < MT; ++t)
        c[t] = __builtin_amdgcn_wmma_f32_16x16x32_f16(false, a2[t], false, b2,
                                                      (short)0, c[t], false,
                                                      false);
    }
  }

  int half = lane >> 4;
  int n = ct * 16 + (lane & 15);
  float bv = bias ? bias[n] : 0.0f;
#pragma unroll
  for (int t = 0; t < MT; ++t) {
#pragma unroll
    for (int j = 0; j < 8; ++j) {  // C/D layout: VGPR j -> row j + 8*half
      int row = (rt0 + t) * 16 + half * 8 + j;
      if (row < N) {
        float v = c[t][j] + bv;
        if (relu) v = fmaxf(v, 0.0f);
        Y[(long long)row * Co + n] = v;
      }
    }
  }
}

// ------------------------------ softmax ------------------------------------
__global__ void softmax32(const float* __restrict__ logits,
                          float* __restrict__ out, int N) {
  int r = blockIdx.x * blockDim.x + threadIdx.x;
  if (r >= N) return;
  const float* p = logits + (long long)r * 32;
  float mx = -1e30f;
#pragma unroll
  for (int i = 0; i < 32; ++i) mx = fmaxf(mx, p[i]);
  float e[32], s = 0.0f;
#pragma unroll
  for (int i = 0; i < 32; ++i) {
    e[i] = __expf(p[i] - mx);
    s += e[i];
  }
  float inv = 1.0f / s;
  float* o = out + (long long)r * 32;
#pragma unroll
  for (int i = 0; i < 32; ++i) o[i] = e[i] * inv;
}

// ---------------------------------------------------------------------------
extern "C" void kernel_launch(void* const* d_in, const int* in_sizes, int n_in,
                              void* d_out, int out_size, void* d_ws,
                              size_t ws_size, hipStream_t stream) {
  const float*     x    = (const float*)d_in[0];
  const long long* ei   = (const long long*)d_in[1];  // int64 edge_index
  const float*     Wl   = (const float*)d_in[2];
  const float*     bl   = (const float*)d_in[3];
  const float*     Wr   = (const float*)d_in[4];
  const float*     bn1g = (const float*)d_in[5];
  const float*     bn1b = (const float*)d_in[6];
  const float*     bn2g = (const float*)d_in[7];
  const float*     bn2b = (const float*)d_in[8];
  const float*     l1W  = (const float*)d_in[9];
  const float*     l1b  = (const float*)d_in[10];
  const float*     l2W  = (const float*)d_in[11];
  const float*     l2b  = (const float*)d_in[12];
  const float*     oW   = (const float*)d_in[13];
  const float*     ob   = (const float*)d_in[14];

  const int F = 128, C = 128;
  const int N = in_sizes[0] / F;
  const int E = in_sizes[1] / 2;
  const long long* src = ei;
  const long long* dst = ei + E;

  const int rtTotal = (N + 15) / 16;
  const int gy = (rtTotal + MT - 1) / MT;
  const int rtPad = gy * MT;           // padded row-tile count
  const size_t NP = (size_t)rtPad * 16;  // padded row count for packed bufs

  // ---- workspace layout --------------------------------------------------
  float* xf    = (float*)d_ws;            // N*128 f32 (scatter source)
  float* aggr  = xf + (size_t)N * 128;    // N*128 f32 (reused for logits)
  float* ybuf  = aggr + (size_t)N * 128;  // N*256 f32
  float* deg   = ybuf + (size_t)N * 256;  // N
  float* stats = deg + N;                 // 512
  char* hraw = (char*)(stats + 512);
  hraw = (char*)(((uintptr_t)hraw + 63) & ~(uintptr_t)63);
  _Float16* xh    = (_Float16*)hraw;      // NP*128 (padded)
  _Float16* aggrh = xh + NP * 128;        // NP*128 (padded)
  _Float16* yh    = aggrh + NP * 128;     // NP*256 (padded)
  _Float16* wlh   = yh + NP * 256;        // 3*128*128
  _Float16* wrh   = wlh + 3 * 128 * 128;  // 3*128*128
  _Float16* l1h   = wrh + 3 * 128 * 128;  // 256*128
  _Float16* l2h   = l1h + 256 * 128;      // 128*256
  _Float16* ohh   = l2h + 128 * 256;      // 32*128

  const int TB = 256;
  auto blocks = [](long long n, int tb) {
    long long b = (n + tb - 1) / tb;
    return (int)(b > 65535 ? 65535 : b);
  };
  const _Float16* nh = nullptr;
  const float* nf = nullptr;

  // ---- zero the pad tails of packed activation buffers (tiny) ------------
  const long long padT = rtPad - rtTotal;  // pad row tiles
  if (padT > 0) {
    // per-rt fragment block = KC*512 halfs; zero as f32 (counts are even)
    long long p128 = padT * 4 * 512 / 2;   // K=128 -> KC=4
    long long p256 = padT * 8 * 512 / 2;   // K=256 -> KC=8
    hipLaunchKernelGGL(fill_f32, dim3(blocks(p128, TB)), dim3(TB), 0, stream,
                       (float*)(xh + (size_t)rtTotal * 4 * 512), 0.0f, p128);
    hipLaunchKernelGGL(fill_f32, dim3(blocks(p128, TB)), dim3(TB), 0, stream,
                       (float*)(aggrh + (size_t)rtTotal * 4 * 512), 0.0f, p128);
    hipLaunchKernelGGL(fill_f32, dim3(blocks(p256, TB)), dim3(TB), 0, stream,
                       (float*)(yh + (size_t)rtTotal * 8 * 512), 0.0f, p256);
  }

  // ---- pack all weights into B-fragment layout (tiny, once per call) -----
  hipLaunchKernelGGL(pack_b, dim3(blocks(3LL * C * C, TB)), dim3(TB), 0,
                     stream, Wl, wlh, 3 * C, C);  // 3 stacked 128x128 mats
  hipLaunchKernelGGL(pack_b, dim3(blocks(3LL * C * C, TB)), dim3(TB), 0,
                     stream, Wr, wrh, 3 * C, C);
  hipLaunchKernelGGL(pack_b, dim3(blocks(256LL * 128, TB)), dim3(TB), 0,
                     stream, l1W, l1h, 256, 128);
  hipLaunchKernelGGL(pack_b, dim3(blocks(128LL * 256, TB)), dim3(TB), 0,
                     stream, l2W, l2h, 128, 256);
  hipLaunchKernelGGL(pack_b, dim3(blocks(32LL * 128, TB)), dim3(TB), 0,
                     stream, oW, ohh, 32, 128);

  // ---- degrees: deg[i] = 1 / max(indeg, 1) -------------------------------
  hipLaunchKernelGGL(fill_f32, dim3(blocks(N, TB)), dim3(TB), 0, stream, deg,
                     0.0f, (long long)N);
  hipLaunchKernelGGL(deg_kernel, dim3((E + TB - 1) / TB), dim3(TB), 0, stream,
                     dst, deg, E);
  hipLaunchKernelGGL(rdeg_kernel, dim3((N + TB - 1) / TB), dim3(TB), 0, stream,
                     deg, N);

  // ---- BN1: x -> xf (f32) + xh (packed f16) ------------------------------
  hipLaunchKernelGGL(fill_f32, dim3(2), dim3(TB), 0, stream, stats, 0.0f,
                     (long long)2 * F);
  hipLaunchKernelGGL(bn_stats, dim3(512), dim3(F), 0, stream, x, stats, N, F);
  hipLaunchKernelGGL(bn_apply_pack, dim3(blocks((long long)N * F, TB)),
                     dim3(TB), 0, stream, x, stats, bn1g, bn1b, xf, xh, N, F);

  // ---- 3x SAGEConv + ReLU + BN2 -----------------------------------------
  for (int l = 0; l < 3; ++l) {
    hipLaunchKernelGGL(fill_f32, dim3(blocks((long long)N * C, TB)), dim3(TB),
                       0, stream, aggr, 0.0f, (long long)N * C);
    hipLaunchKernelGGL(scatter_accum,
                       dim3(blocks((long long)E * (C >> 2), TB)), dim3(TB), 0,
                       stream, xf, src, dst, aggr, E, C);
    // pack mean-aggregate: aggrh = frag( aggr * (1/deg) )
    hipLaunchKernelGGL(pack_a, dim3(blocks((long long)N * C, TB)), dim3(TB), 0,
                       stream, aggr, deg, aggrh, N, C);
    // ybuf = relu( aggrh @ Wl^T + bl + xh @ Wr^T )
    hipLaunchKernelGGL(wmma_gemm<true>, dim3(C / 16, gy), dim3(32), 0, stream,
                       aggrh, wlh + (size_t)l * C * C, xh,
                       wrh + (size_t)l * C * C, bl + (size_t)l * C, ybuf, N, C,
                       C, 1);
    // BN2: ybuf -> xf + xh
    hipLaunchKernelGGL(fill_f32, dim3(2), dim3(TB), 0, stream, stats, 0.0f,
                       (long long)2 * C);
    hipLaunchKernelGGL(bn_stats, dim3(512), dim3(C), 0, stream, ybuf, stats, N,
                       C);
    hipLaunchKernelGGL(bn_apply_pack, dim3(blocks((long long)N * C, TB)),
                       dim3(TB), 0, stream, ybuf, stats, bn2g, bn2b, xf, xh, N,
                       C);
  }

  // ---- MLP head ----------------------------------------------------------
  // ybuf[N,256] = relu(xh @ l1W^T + l1b)
  hipLaunchKernelGGL(wmma_gemm<false>, dim3(256 / 16, gy), dim3(32), 0, stream,
                     xh, l1h, nh, nh, l1b, ybuf, N, 128, 256, 1);
  hipLaunchKernelGGL(pack_a, dim3(blocks((long long)N * 256, TB)), dim3(TB), 0,
                     stream, ybuf, nf, yh, N, 256);
  // xf[N,128] = relu(yh @ l2W^T + l2b)
  hipLaunchKernelGGL(wmma_gemm<false>, dim3(128 / 16, gy), dim3(32), 0, stream,
                     yh, l2h, nh, nh, l2b, xf, N, 256, 128, 1);
  hipLaunchKernelGGL(pack_a, dim3(blocks((long long)N * 128, TB)), dim3(TB), 0,
                     stream, xf, nf, xh, N, 128);
  // aggr[N,32] = xh @ oW^T + ob   (reuse aggr as logits buffer)
  hipLaunchKernelGGL(wmma_gemm<false>, dim3(32 / 16, gy), dim3(32), 0, stream,
                     xh, ohh, nh, nh, ob, aggr, N, 128, 32, 0);

  // ---- softmax -> d_out --------------------------------------------------
  hipLaunchKernelGGL(softmax32, dim3((N + TB - 1) / TB), dim3(TB), 0, stream,
                     aggr, (float*)d_out, N);
}
